// FourPosFusionEmbedding_23141283790923
// MI455X (gfx1250) — matrix operand order repaired and checked by reading, not compiled, not run on hardware
//
#include <hip/hip_runtime.h>

typedef __attribute__((ext_vector_type(16))) _Float16     v16h;
typedef __attribute__((ext_vector_type(8)))  float        v8f;
typedef __attribute__((ext_vector_type(4)))  int          v4i;
typedef __attribute__((ext_vector_type(4)))  unsigned int v4u;

#define BATCH   2
#define SEQ     384
#define HID     128
#define FDIM    512          // 4*HID
#define TABLE   (2 * SEQ + 1)
#define MTILE   64           // output rows per block
#define KCHUNK  128          // K chunk == one table width
#define LDSTR   136          // padded LDS row stride (halves); 272B, 16B-aligned

#define HAS_ASYNC_LDS __has_builtin(__builtin_amdgcn_global_load_async_to_lds_b128)

// ---------------------------------------------------------------------------
// Prologue: one-time f32 -> f16 conversion of W and the 4 PE tables into d_ws.
// Layout in ws (halves): Wh[128][512] at 0, then T16[4][TABLE][128].
// ---------------------------------------------------------------------------
__global__ __launch_bounds__(256)
void fourpos_cvt_f16_kernel(const float* __restrict__ W,
                            const float* __restrict__ pe_ss,
                            const float* __restrict__ pe_se,
                            const float* __restrict__ pe_es,
                            const float* __restrict__ pe_ee,
                            _Float16*    __restrict__ Wh,
                            _Float16*    __restrict__ T16)
{
    const int i   = blockIdx.x * 256 + threadIdx.x;
    const int seg = blockIdx.y;
    if (seg == 0) {
        if (i < HID * FDIM) Wh[i] = (_Float16)W[i];
    } else {
        const float* src = (seg == 1) ? pe_ss : (seg == 2) ? pe_se
                         : (seg == 3) ? pe_es : pe_ee;
        if (i < TABLE * HID) T16[(seg - 1) * (TABLE * HID) + i] = (_Float16)src[i];
    }
}

// ---------------------------------------------------------------------------
// 16-byte stage helper: async DMA to LDS when available, else vector copy.
// ---------------------------------------------------------------------------
__device__ __forceinline__
void stage16(const _Float16* gsrc, _Float16* ldst)
{
#if HAS_ASYNC_LDS
    __builtin_amdgcn_global_load_async_to_lds_b128(
        (__attribute__((address_space(1))) v4i*)gsrc,
        (__attribute__((address_space(3))) v4i*)ldst,
        /*offset=*/0, /*cpol=*/0);
#else
    *(v4u*)ldst = *(const v4u*)gsrc;
#endif
}

__device__ __forceinline__ void stage_wait()
{
#if HAS_ASYNC_LDS
  #if __has_builtin(__builtin_amdgcn_s_wait_asynccnt)
    __builtin_amdgcn_s_wait_asynccnt(0);
  #else
    asm volatile("s_wait_asynccnt 0x0" ::: "memory");
  #endif
#endif
}

// ---------------------------------------------------------------------------
// Main kernel: per block, one 64(M=k rows) x 128(N=h) output tile for (b,q).
// K=512 processed as 4 chunks of 128; chunk c maps exactly to table c.
// ---------------------------------------------------------------------------
__global__ __launch_bounds__(256)
void fourpos_fusion_wmma_kernel(const int*      __restrict__ pos_s,
                                const int*      __restrict__ pos_e,
                                const _Float16* __restrict__ T16,
                                const _Float16* __restrict__ Wh,
                                const float*    __restrict__ bias,
                                float*          __restrict__ out)
{
    __shared__ _Float16 As[MTILE][LDSTR];   // gathered pe rows (f16), one K-chunk
    __shared__ _Float16 Ws[HID][LDSTR];     // Ws[n][kk] = Wh[n][c*128+kk]

    const int tid = threadIdx.x;
    const int bid = blockIdx.x;

    const int mt = bid % (SEQ / MTILE);           // which 64-row band of k
    const int q  = (bid / (SEQ / MTILE)) % SEQ;
    const int b  = bid / ((SEQ / MTILE) * SEQ);

    const int ps_q = pos_s[b * SEQ + q];
    const int pe_q = pos_e[b * SEQ + q];

    const int wave   = tid >> 5;        // 0..7
    const int lane   = tid & 31;
    const int m_sub  = wave >> 1;       // 0..3 : 16-row sub-band
    const int n_half = wave & 1;        // 0..1 : 64-col half
    const int group  = lane >> 4;       // lane group 0/1
    const int l16    = lane & 15;

    v8f acc[4];
    #pragma unroll
    for (int t = 0; t < 4; ++t) { v8f z = {}; acc[t] = z; }

    #pragma unroll
    for (int c = 0; c < 4; ++c) {
        // ---- stage W chunk: 128 rows x 128 halves = 2048 x 16B, 8 passes
        #pragma unroll
        for (int it = 0; it < 8; ++it) {
            int xv = it * 256 + tid;
            int n  = xv >> 4;
            int x  = (xv & 15) * 8;
            stage16(&Wh[n * FDIM + c * KCHUNK + x], &Ws[n][x]);
        }
        // ---- gather A chunk: 64 rows x 128 halves = 1024 x 16B, 4 passes
        #pragma unroll
        for (int it = 0; it < 4; ++it) {
            int xv  = it * 256 + tid;
            int m   = xv >> 4;
            int x   = (xv & 15) * 8;
            int kr  = mt * MTILE + m;
            int aq  = (c < 2) ? ps_q : pe_q;
            int ak  = ((c & 1) == 0) ? pos_s[b * SEQ + kr] : pos_e[b * SEQ + kr];
            int idx = aq - ak + SEQ;                 // [0, 2*SEQ]
            stage16(&T16[(c * TABLE + idx) * HID + x], &As[m][x]);
        }
        stage_wait();
        __syncthreads();

        // ---- 4 K-steps of 32 over this chunk
        #pragma unroll
        for (int ks = 0; ks < KCHUNK; ks += 32) {
            // A fragment: 16x32 f16, ISA layout:
            //  lane<16: VGPR j<4 -> K=2j,2j+1 ; j>=4 -> K=16+2(j-4),...
            //  lane>=16: same +8
            union { v16h v; unsigned int u[8]; } afrag;
            #pragma unroll
            for (int j = 0; j < 8; ++j) {
                int kl = ((j < 4) ? (2 * j) : (16 + 2 * (j - 4))) + group * 8;
                afrag.u[j] = *(const unsigned int*)&As[m_sub * 16 + l16][ks + kl];
            }

            #pragma unroll
            for (int t = 0; t < 4; ++t) {
                int n0 = n_half * 64 + t * 16;
                // B fragment: 32x16 f16, same K layout as A:
                // VGPR j<4 -> K=2j,2j+1 ; j>=4 -> K=16+2(j-4),... ; +8 for
                // lanes>=16; column n = n0 + l16.
                union { v16h v; unsigned int u[8]; } bfrag;
                #pragma unroll
                for (int j = 0; j < 8; ++j) {
                    int kl = ((j < 4) ? (2 * j) : (16 + 2 * (j - 4))) + group * 8;
                    bfrag.u[j] = *(const unsigned int*)&Ws[n0 + l16][ks + kl];
                }
                acc[t] = __builtin_amdgcn_wmma_f32_16x16x32_f16(
                    /*neg_a=*/false, afrag.v, /*neg_b=*/false, bfrag.v,
                    /*c_mod=*/(short)0, acc[t], /*reuse_a=*/false, /*reuse_b=*/false);
            }
        }
        __syncthreads();
    }

    // ---- epilogue: bias + ReLU + non-temporal store
    //      (C/D layout: VGPR r -> M = group*8 + r, N = l16 within tile)
    #pragma unroll
    for (int t = 0; t < 4; ++t) {
        int   n  = n_half * 64 + t * 16 + l16;
        float bv = bias[n];
        #pragma unroll
        for (int r = 0; r < 8; ++r) {
            int m  = group * 8 + r;
            int kr = mt * MTILE + m_sub * 16 + m;
            float v = acc[t][r] + bv;
            v = v > 0.0f ? v : 0.0f;
            __builtin_nontemporal_store(v, &out[(((size_t)b * SEQ + q) * SEQ + kr) * HID + n]);
        }
    }
}

extern "C" void kernel_launch(void* const* d_in, const int* in_sizes, int n_in,
                              void* d_out, int out_size, void* d_ws, size_t ws_size,
                              hipStream_t stream) {
    (void)in_sizes; (void)n_in; (void)out_size; (void)ws_size;

    const int*   pos_s = (const int*)d_in[0];
    const int*   pos_e = (const int*)d_in[1];
    const float* pe_ss = (const float*)d_in[2];
    const float* pe_se = (const float*)d_in[3];
    const float* pe_es = (const float*)d_in[4];
    const float* pe_ee = (const float*)d_in[5];
    const float* W     = (const float*)d_in[6];
    const float* bias  = (const float*)d_in[7];
    float*       out   = (float*)d_out;

    _Float16* Wh  = (_Float16*)d_ws;                 // 128*512 halves
    _Float16* T16 = Wh + HID * FDIM;                 // 4*TABLE*128 halves

    // 1) one-time f16 conversion of W + tables into scratch
    {
        dim3 grid((TABLE * HID + 255) / 256, 5);     // y=0: W, y=1..4: tables
        fourpos_cvt_f16_kernel<<<grid, 256, 0, stream>>>(
            W, pe_ss, pe_se, pe_es, pe_ee, Wh, T16);
    }
    // 2) WMMA main kernel
    {
        dim3 grid(BATCH * SEQ * (SEQ / MTILE));      // 2*384*6 = 4608 blocks
        fourpos_fusion_wmma_kernel<<<grid, 256, 0, stream>>>(
            pos_s, pos_e, T16, Wh, bias, out);
    }
}
